// TRecTransformer_27006754357284
// MI455X (gfx1250) — matrix operand order repaired and checked by reading, not compiled, not run on hardware
//
#include <hip/hip_runtime.h>
#include <cstdint>
#include <cstddef>

// ---------------------------------------------------------------------------
// CDNA5 (gfx1250) wave32 WMMA types
// ---------------------------------------------------------------------------
typedef __attribute__((ext_vector_type(16))) __bf16 bf16x16;
typedef __attribute__((ext_vector_type(8)))  float  f32x8;

__device__ __forceinline__ f32x8 wmma_bf16(bf16x16 a, bf16x16 b, f32x8 c) {
  // D = A(16x32 bf16) * B(32x16 bf16) + C(16x16 f32)
  return __builtin_amdgcn_wmma_f32_16x16x32_bf16(false, a, false, b, (short)0, c,
                                                 false, false);
}

union frag_u { bf16x16 v; uint4 q[2]; };

#define PI_F 3.14159265358979323846f

// ---------------------------------------------------------------------------
// Weight convert: W[K][N] f32 -> Wt[N][K] bf16 (one-time per launch)
// ---------------------------------------------------------------------------
__global__ void k_wt(const float* __restrict__ W, __bf16* __restrict__ Wt,
                     int K, int N) {
  int idx = blockIdx.x * 256 + threadIdx.x;
  if (idx >= K * N) return;
  int k = idx / N, n = idx % N;
  Wt[(size_t)n * K + k] = (__bf16)W[idx];
}

// ---------------------------------------------------------------------------
// Embedding: z[b,n,c] = (c<128) ? x[b,n,:]@emb_W + emb_b : pe_input[n,c-128]
// ---------------------------------------------------------------------------
__global__ void k_embed(const float* __restrict__ x, const float* __restrict__ pe,
                        const float* __restrict__ eW, const float* __restrict__ eb,
                        float* __restrict__ zf, __bf16* __restrict__ zb) {
  int row = blockIdx.x;        // 0..16383  ([b][n])
  int c   = threadIdx.x;       // 0..255
  int n   = row & 4095;
  float v;
  if (c < 128) v = x[row * 2] * eW[c] + x[row * 2 + 1] * eW[128 + c] + eb[c];
  else         v = pe[n * 128 + (c - 128)];
  size_t o = (size_t)row * 256 + c;
  zf[o] = v;
  zb[o] = (__bf16)v;
}

// Broadcast decoder queries: t[b,l,c] = out_pos_emb[l,c]
__global__ void k_bcast(const float* __restrict__ src, float* __restrict__ dstf,
                        __bf16* __restrict__ dstb, int n) {
  int idx = blockIdx.x * 256 + threadIdx.x;
  if (idx >= n) return;
  float v = src[idx % (8320 * 256)];
  dstf[idx] = v;
  dstb[idx] = (__bf16)v;
}

// ---------------------------------------------------------------------------
// Tiled bf16 WMMA GEMM: C[M,N] = act(A[M,K] @ Wt[N,K]^T + bias)
// block = 128 threads = 4 waves; 128x64 tile; each wave 32 rows x 64 cols;
// K-step 32 -> 8 WMMAs per wave per step, B fragments reused 2x.
// ---------------------------------------------------------------------------
template<bool RELU, bool WF32, bool WBF16>
__global__ __launch_bounds__(128)
void k_gemm(const __bf16* __restrict__ A, const __bf16* __restrict__ Wt,
            const float* __restrict__ bias,
            float* __restrict__ Cf, __bf16* __restrict__ Cb,
            int M, int N, int K) {
  __shared__ __attribute__((aligned(16))) __bf16 As[128 * 40];
  __shared__ __attribute__((aligned(16))) __bf16 Bs[64 * 40];
  const int tid  = threadIdx.x;
  const int wv   = tid >> 5, lane = tid & 31;
  const int l16  = lane & 15, hi = lane >> 4;
  const size_t bm = (size_t)blockIdx.y * 128, bn = (size_t)blockIdx.x * 64;
  f32x8 acc[2][4] = {};
  for (int k0 = 0; k0 < K; k0 += 32) {
    if (k0 + 32 < K) {  // next-tile prefetch -> global_prefetch_b8
      __builtin_prefetch(&A [(bm + tid) * K + k0 + 32], 0, 1);
      __builtin_prefetch(&Wt[(bn + (tid >> 1)) * K + k0 + 32 + (tid & 1) * 16], 0, 1);
    }
    // stage A 128x32 (512 chunks of 8) and B 64x32 (256 chunks of 8), b128 each
#pragma unroll
    for (int i = 0; i < 4; ++i) {
      int c = tid + i * 128;
      int r = c >> 2, k8 = (c & 3) * 8;
      *(uint4*)&As[r * 40 + k8] = *(const uint4*)&A[(bm + r) * K + k0 + k8];
    }
#pragma unroll
    for (int i = 0; i < 2; ++i) {
      int c = tid + i * 128;
      int r = c >> 2, k8 = (c & 3) * 8;
      *(uint4*)&Bs[r * 40 + k8] = *(const uint4*)&Wt[(bn + r) * K + k0 + k8];
    }
    __syncthreads();
    // A fragments (16x32): lane row = l16, K halves selected by hi
    bf16x16 af[2];
#pragma unroll
    for (int s = 0; s < 2; ++s) {
      const __bf16* pr = &As[(wv * 32 + s * 16 + l16) * 40];
      frag_u u;
      u.q[0] = *(const uint4*)(pr + 8 * hi);
      u.q[1] = *(const uint4*)(pr + 16 + 8 * hi);
      af[s] = u.v;
    }
#pragma unroll
    for (int nt = 0; nt < 4; ++nt) {
      // B fragment (32x16): lane col = l16, K halves selected by hi (transposed tile)
      const __bf16* pb = &Bs[(nt * 16 + l16) * 40];
      frag_u u;
      u.q[0] = *(const uint4*)(pb + 8 * hi);
      u.q[1] = *(const uint4*)(pb + 16 + 8 * hi);
      acc[0][nt] = wmma_bf16(af[0], u.v, acc[0][nt]);
      acc[1][nt] = wmma_bf16(af[1], u.v, acc[1][nt]);
    }
    __syncthreads();
  }
#pragma unroll
  for (int s = 0; s < 2; ++s)
#pragma unroll
    for (int nt = 0; nt < 4; ++nt) {
      int col  = (int)bn + nt * 16 + l16;
      float bv = bias[col];
#pragma unroll
      for (int r = 0; r < 8; ++r) {
        size_t row = bm + wv * 32 + s * 16 + r + 8 * hi;
        float o = acc[s][nt][r] + bv;
        if (RELU)  o = fmaxf(o, 0.f);
        if (WF32)  Cf[row * N + col] = o;
        if (WBF16) Cb[row * N + col] = (__bf16)o;
      }
    }
}

// ---------------------------------------------------------------------------
// elu(x)+1 feature map, in place on bf16 buffer
// ---------------------------------------------------------------------------
__global__ void k_elu1(__bf16* __restrict__ q, int n) {
  int i = blockIdx.x * 256 + threadIdx.x;
  if (i >= n) return;
  float x = (float)q[i];
  q[i] = (__bf16)(x > 0.f ? x + 1.f : __expf(x));
}

// ---------------------------------------------------------------------------
// KV[b,h] = Kf^T @ V  (32x32 per head), WMMA over sequence dim.
// K,V staged TRANSPOSED in LDS so fragments are contiguous b128 loads.
// Output stored transposed: kvt[b,h][m][d].
// grid = B*H blocks of 128 threads (4 waves -> 4 16x16 tiles)
// ---------------------------------------------------------------------------
__global__ __launch_bounds__(128)
void k_kv(const __bf16* __restrict__ kf, const __bf16* __restrict__ v,
          __bf16* __restrict__ kvt, int Lkv) {
  int bh = blockIdx.x, b = bh >> 3, h = bh & 7;
  __shared__ __attribute__((aligned(16))) __bf16 Kt[32][40];  // [d][l]
  __shared__ __attribute__((aligned(16))) __bf16 Vt[32][40];  // [m][l]
  int tid = threadIdx.x, wv = tid >> 5, lane = tid & 31;
  int dt = wv >> 1, mt = wv & 1;
  int l16 = lane & 15, hi = lane >> 4;
  f32x8 acc = {};
  for (int l0 = 0; l0 < Lkv; l0 += 32) {
    int lr = tid >> 2, c8 = (tid & 3) * 8;
    size_t g = ((size_t)(b * Lkv + l0 + lr)) * 256 + h * 32 + c8;
    union { uint4 q; __bf16 e[8]; } kq, vq;
    kq.q = *(const uint4*)&kf[g];
    vq.q = *(const uint4*)&v[g];
#pragma unroll
    for (int j = 0; j < 8; ++j) { Kt[c8 + j][lr] = kq.e[j]; Vt[c8 + j][lr] = vq.e[j]; }
    __syncthreads();
    frag_u ua;  // A = Kf^T : A[d][l]
    {
      const __bf16* pk = &Kt[dt * 16 + l16][0];
      ua.q[0] = *(const uint4*)(pk + 8 * hi);
      ua.q[1] = *(const uint4*)(pk + 16 + 8 * hi);
    }
    frag_u ub;  // B = V : B[l][m] staged as Vt[m][l]
    {
      const __bf16* pv = &Vt[mt * 16 + l16][0];
      ub.q[0] = *(const uint4*)(pv + 8 * hi);
      ub.q[1] = *(const uint4*)(pv + 16 + 8 * hi);
    }
    acc = wmma_bf16(ua.v, ub.v, acc);
    __syncthreads();
  }
#pragma unroll
  for (int r = 0; r < 8; ++r) {
    int d = dt * 16 + r + 8 * hi, m = mt * 16 + l16;
    kvt[((size_t)bh * 32 + m) * 32 + d] = (__bf16)acc[r];   // transposed at rest
  }
}

// Ksum[b,h,d] = sum_l Kf[b,l,h,d]
__global__ __launch_bounds__(128)
void k_ksum(const __bf16* __restrict__ kf, float* __restrict__ ksum, int Lkv) {
  int bh = blockIdx.x, b = bh >> 3, h = bh & 7;
  int tid = threadIdx.x, d = tid & 31, part = tid >> 5;
  float s = 0.f;
  for (int l = part; l < Lkv; l += 4)
    s += (float)kf[((size_t)(b * Lkv + l)) * 256 + h * 32 + d];
  __shared__ float red[128];
  red[tid] = s;
  __syncthreads();
  if (tid < 32) ksum[bh * 32 + d] = red[d] + red[32 + d] + red[64 + d] + red[96 + d];
}

// ---------------------------------------------------------------------------
// out[l,h,:] = (Qf[l,h,:] @ KV[b,h]) / (Qf[l,h,:].Ksum[b,h] + eps)
// KV arrives transposed (kvt[m][d]) -> contiguous b128 fragment loads.
// grid = (Lq/64, B*H), 128 threads; Q tile 64x32.
// ---------------------------------------------------------------------------
__global__ __launch_bounds__(128)
void k_attn_out(const __bf16* __restrict__ qf, const __bf16* __restrict__ kvt,
                const float* __restrict__ ksum, __bf16* __restrict__ outb, int Lq) {
  int bh = blockIdx.y, b = bh >> 3, h = bh & 7;
  int l0 = blockIdx.x * 64;
  __shared__ __attribute__((aligned(16))) __bf16 KVs[32][40];  // [m][d]
  __shared__ __attribute__((aligned(16))) __bf16 Qs[64][40];   // [l][d]
  __shared__ float zs[64];
  int tid = threadIdx.x, wv = tid >> 5, lane = tid & 31;
  int l16 = lane & 15, hi = lane >> 4;
  {
    int r = tid >> 2, c8 = (tid & 3) * 8;   // 128 chunks of 8
    *(uint4*)&KVs[r][c8] = *(const uint4*)&kvt[((size_t)bh * 32 + r) * 32 + c8];
  }
#pragma unroll
  for (int i = 0; i < 2; ++i) {
    int c = tid + i * 128;                  // 256 chunks of 8
    int r = c >> 2, k8 = (c & 3) * 8;
    *(uint4*)&Qs[r][k8] =
        *(const uint4*)&qf[((size_t)(b * Lq + l0 + r)) * 256 + h * 32 + k8];
  }
  __syncthreads();
  if (tid < 64) {
    float s = 0.f;
#pragma unroll
    for (int d = 0; d < 32; ++d) s += (float)Qs[tid][d] * ksum[bh * 32 + d];
    zs[tid] = 1.f / (s + 1e-6f);
  }
  __syncthreads();
  frag_u ua;
  {
    const __bf16* pq = &Qs[wv * 16 + l16][0];
    ua.q[0] = *(const uint4*)(pq + 8 * hi);
    ua.q[1] = *(const uint4*)(pq + 16 + 8 * hi);
  }
#pragma unroll
  for (int mt = 0; mt < 2; ++mt) {
    frag_u ub;   // B[k=d][n=m] staged as KVs[m][d]
    const __bf16* pb = &KVs[mt * 16 + l16][0];
    ub.q[0] = *(const uint4*)(pb + 8 * hi);
    ub.q[1] = *(const uint4*)(pb + 16 + 8 * hi);
    f32x8 acc = {};
    acc = wmma_bf16(ua.v, ub.v, acc);
#pragma unroll
    for (int r = 0; r < 8; ++r) {
      int rl = wv * 16 + r + 8 * hi;
      size_t row = (size_t)(b * Lq + l0 + rl);
      outb[row * 256 + h * 32 + mt * 16 + l16] = (__bf16)(acc[r] * zs[rl]);
    }
  }
}

// ---------------------------------------------------------------------------
// Fused residual add + LayerNorm(256) -> f32 stream + bf16 copy
// block = 256 threads = 8 waves, one wave32 per row (8 elems/lane)
// ---------------------------------------------------------------------------
__global__ __launch_bounds__(256)
void k_add_ln(const float* __restrict__ x, const float* __restrict__ y,
              const float* __restrict__ g, const float* __restrict__ be,
              float* __restrict__ of, __bf16* __restrict__ ob, int M) {
  int wv = threadIdx.x >> 5, lane = threadIdx.x & 31;
  int row = blockIdx.x * 8 + wv;
  if (row >= M) return;
  const float* px = x + (size_t)row * 256;
  const float* py = y + (size_t)row * 256;
  float v[8], s = 0.f, sq = 0.f;
#pragma unroll
  for (int j = 0; j < 8; ++j) {
    v[j] = px[lane + j * 32] + py[lane + j * 32];
    s += v[j]; sq += v[j] * v[j];
  }
#pragma unroll
  for (int o = 16; o > 0; o >>= 1) {
    s  += __shfl_xor(s, o, 32);
    sq += __shfl_xor(sq, o, 32);
  }
  float mean = s * (1.f / 256.f);
  float var  = sq * (1.f / 256.f) - mean * mean;
  float inv  = rsqrtf(var + 1e-5f);
#pragma unroll
  for (int j = 0; j < 8; ++j) {
    int c = lane + j * 32;
    float o = (v[j] - mean) * inv * g[c] + be[c];
    of[(size_t)row * 256 + c] = o;
    ob[(size_t)row * 256 + c] = (__bf16)o;
  }
}

// ---------------------------------------------------------------------------
// pred head (N=2), dft build/scatter, attenuation, naive iFFT, convs, BN
// ---------------------------------------------------------------------------
__global__ void k_pred(const __bf16* __restrict__ t, const float* __restrict__ W,
                       const float* __restrict__ b, float* __restrict__ out, int M) {
  int row = blockIdx.x * 256 + threadIdx.x;
  if (row >= M) return;
  float a0 = 0.f, a1 = 0.f;
  for (int c = 0; c < 256; ++c) {
    float tv = (float)t[(size_t)row * 256 + c];
    a0 += tv * W[c * 2 + 0];
    a1 += tv * W[c * 2 + 1];
  }
  out[row * 2 + 0] = a0 + b[0];
  out[row * 2 + 1] = a1 + b[1];
}

__global__ void k_dft(const float* __restrict__ yhat, const int* __restrict__ coords,
                      const float* __restrict__ mn, const float* __restrict__ mx,
                      float2* __restrict__ dft, int n) {
  int idx = blockIdx.x * 256 + threadIdx.x;
  if (idx >= n) return;
  int b = idx / 8320, j = idx % 8320;
  float amp = __expf((yhat[idx * 2] + 1.f) * 0.5f * (mx[0] - mn[0]) + mn[0]) - 1.f;
  float s, c;
  __sincosf(yhat[idx * 2 + 1] * PI_F, &s, &c);
  dft[b * 8320 + coords[j]] = make_float2(amp * c, amp * s);
}

__global__ void k_atten(float2* __restrict__ dft, const float* __restrict__ att, int n) {
  int idx = blockIdx.x * 256 + threadIdx.x;
  if (idx >= n) return;
  float a = att[idx % 8320];
  dft[idx].x *= a;
  dft[idx].y *= a;
}

// inverse DFT along y (complex, length 128)
__global__ void k_ifft_y(const float2* __restrict__ dft, float2* __restrict__ tm) {
  int idx = blockIdx.x * 256 + threadIdx.x;
  if (idx >= 4 * 128 * 65) return;
  int x = idx % 65, y2 = (idx / 65) % 128, b = idx / (65 * 128);
  float re = 0.f, im = 0.f;
  for (int y = 0; y < 128; ++y) {
    float s, c;
    __sincosf((float)((y * y2) & 127) * (2.f * PI_F / 128.f), &s, &c);
    float2 v = dft[(b * 128 + y) * 65 + x];
    re += v.x * c - v.y * s;
    im += v.x * s + v.y * c;
  }
  tm[idx] = make_float2(re, im);
}

// inverse rFFT along x (hermitian) + roll(S/2,S/2)
__global__ void k_ifft_x(const float2* __restrict__ tm, float* __restrict__ img) {
  int idx = blockIdx.x * 256 + threadIdx.x;
  if (idx >= 4 * 128 * 128) return;
  int x2 = idx & 127, y2 = (idx >> 7) & 127, b = idx >> 14;
  float acc = 0.f;
  for (int x = 0; x < 65; ++x) {
    float w = (x == 0 || x == 64) ? 1.f : 2.f;
    float s, c;
    __sincosf((float)((x * x2) & 127) * (2.f * PI_F / 128.f), &s, &c);
    float2 v = tm[(b * 128 + y2) * 65 + x];
    acc += w * (v.x * c - v.y * s);
  }
  int yr = (y2 + 64) & 127, xr = (x2 + 64) & 127;
  img[(b << 14) + (yr << 7) + xr] = acc * (1.f / 16384.f);
}

__global__ void k_conv1(const float* __restrict__ img, const float* __restrict__ w,
                        const float* __restrict__ b, float* __restrict__ out) {
  int idx = blockIdx.x * 256 + threadIdx.x;   // [b][c][y][x]
  if (idx >= (1 << 21)) return;
  int x = idx & 127, y = (idx >> 7) & 127, c = (idx >> 14) & 31, bb = idx >> 19;
  float acc = b[c];
#pragma unroll
  for (int dy = -1; dy <= 1; ++dy)
#pragma unroll
    for (int dx = -1; dx <= 1; ++dx) {
      int yy = y + dy, xx = x + dx;
      if (yy >= 0 && yy < 128 && xx >= 0 && xx < 128)
        acc += img[(bb << 14) + (yy << 7) + xx] * w[c * 9 + (dy + 1) * 3 + (dx + 1)];
    }
  out[idx] = fmaxf(acc, 0.f);
}

__global__ __launch_bounds__(256)
void k_bnstats(const float* __restrict__ buf, float* __restrict__ mv) {
  int c = blockIdx.x;
  float s = 0.f, sq = 0.f;
  for (int i = threadIdx.x; i < 65536; i += 256) {
    int b = i >> 14, p = i & 16383;
    float v = buf[((size_t)(b * 32 + c) << 14) + p];
    s += v; sq += v * v;
  }
  __shared__ float rs[256], rq[256];
  rs[threadIdx.x] = s; rq[threadIdx.x] = sq;
  __syncthreads();
  for (int o = 128; o > 0; o >>= 1) {
    if (threadIdx.x < o) { rs[threadIdx.x] += rs[threadIdx.x + o]; rq[threadIdx.x] += rq[threadIdx.x + o]; }
    __syncthreads();
  }
  if (threadIdx.x == 0) {
    float m = rs[0] * (1.f / 65536.f);
    mv[c] = m;
    mv[32 + c] = rq[0] * (1.f / 65536.f) - m * m;
  }
}

__global__ void k_bnapply(float* __restrict__ buf, const float* __restrict__ mv,
                          const float* __restrict__ g, const float* __restrict__ b) {
  int idx = blockIdx.x * 256 + threadIdx.x;
  if (idx >= (1 << 21)) return;
  int c = (idx >> 14) & 31;
  buf[idx] = (buf[idx] - mv[c]) * rsqrtf(mv[32 + c] + 1e-5f) * g[c] + b[c];
}

__global__ void k_conv2(const float* __restrict__ in, const float* __restrict__ w,
                        const float* __restrict__ b, float* __restrict__ out) {
  int idx = blockIdx.x * 256 + threadIdx.x;
  if (idx >= (1 << 21)) return;
  int x = idx & 127, y = (idx >> 7) & 127, c = (idx >> 14) & 31, bb = idx >> 19;
  float acc = b[c];
  for (int ci = 0; ci < 32; ++ci)
#pragma unroll
    for (int dy = -1; dy <= 1; ++dy)
#pragma unroll
      for (int dx = -1; dx <= 1; ++dx) {
        int yy = y + dy, xx = x + dx;
        if (yy >= 0 && yy < 128 && xx >= 0 && xx < 128)
          acc += in[((size_t)(bb * 32 + ci) << 14) + (yy << 7) + xx] *
                 w[((c * 32 + ci) * 3 + (dy + 1)) * 3 + (dx + 1)];
      }
  out[idx] = fmaxf(acc, 0.f);
}

__global__ void k_conv3(const float* __restrict__ in, const float* __restrict__ img,
                        const float* __restrict__ w, const float* __restrict__ b,
                        float* __restrict__ out) {
  int idx = blockIdx.x * 256 + threadIdx.x;
  if (idx >= 4 * 16384) return;
  int p = idx & 16383, bb = idx >> 14;
  float acc = b[0] + img[idx];
  for (int c = 0; c < 32; ++c)
    acc += in[((size_t)(bb * 32 + c) << 14) + p] * w[c];
  out[idx] = acc;
}

// ---------------------------------------------------------------------------
// Host orchestration
// ---------------------------------------------------------------------------
extern "C" void kernel_launch(void* const* d_in, const int* in_sizes, int n_in,
                              void* d_out, int out_size, void* d_ws, size_t ws_size,
                              hipStream_t stream) {
  (void)in_sizes; (void)n_in; (void)out_size; (void)ws_size;
  const float* x        = (const float*)d_in[0];
  const float* pe_input = (const float*)d_in[1];
  const float* out_pos  = (const float*)d_in[2];
  const float* mag_min  = (const float*)d_in[3];
  const float* mag_max  = (const float*)d_in[4];
  const float* atten    = (const float*)d_in[5];
  const float* emb_W    = (const float*)d_in[6];
  const float* emb_b    = (const float*)d_in[7];
  const float* enc_W    = (const float*)d_in[8];
  const float* enc_b    = (const float*)d_in[9];
  const float* enc_f1   = (const float*)d_in[10];
  const float* enc_f1b  = (const float*)d_in[11];
  const float* enc_f2   = (const float*)d_in[12];
  const float* enc_f2b  = (const float*)d_in[13];
  const float* enc_lng  = (const float*)d_in[14];
  const float* enc_lnb  = (const float*)d_in[15];
  const float* dsa_W    = (const float*)d_in[16];
  const float* dsa_b    = (const float*)d_in[17];
  const float* dca_W    = (const float*)d_in[18];
  const float* dca_b    = (const float*)d_in[19];
  const float* dec_f1   = (const float*)d_in[20];
  const float* dec_f1b  = (const float*)d_in[21];
  const float* dec_f2   = (const float*)d_in[22];
  const float* dec_f2b  = (const float*)d_in[23];
  const float* dec_lng  = (const float*)d_in[24];
  const float* dec_lnb  = (const float*)d_in[25];
  const float* pred_W   = (const float*)d_in[26];
  const float* pred_b   = (const float*)d_in[27];
  const float* c1w = (const float*)d_in[28];
  const float* c1b = (const float*)d_in[29];
  const float* bn1g = (const float*)d_in[30];
  const float* bn1b = (const float*)d_in[31];
  const float* c2w = (const float*)d_in[32];
  const float* c2b = (const float*)d_in[33];
  const float* bn2g = (const float*)d_in[34];
  const float* bn2b = (const float*)d_in[35];
  const float* c3w = (const float*)d_in[36];
  const float* c3b = (const float*)d_in[37];
  const int* coords = (const int*)d_in[38];
  float* out = (float*)d_out;

  const int Bz = 4, Le = 4096, Ld = 8320;
  const int Me = Bz * Le, Md = Bz * Ld;

  char* wp = (char*)d_ws;
  auto carve = [&](size_t bytes) -> void* {
    void* p = (void*)wp;
    wp += (bytes + 255) & ~(size_t)255;
    return p;
  };
  __bf16 *wtEnc[4][4], *wtEncF1[4], *wtEncF2[4];
  __bf16 *wtSa[4][4], *wtCa[4][4], *wtDecF1[4], *wtDecF2[4];
  for (int i = 0; i < 4; ++i) {
    for (int j = 0; j < 4; ++j) wtEnc[i][j] = (__bf16*)carve(65536 * 2);
    wtEncF1[i] = (__bf16*)carve(262144 * 2);
    wtEncF2[i] = (__bf16*)carve(262144 * 2);
    for (int j = 0; j < 4; ++j) wtSa[i][j] = (__bf16*)carve(65536 * 2);
    for (int j = 0; j < 4; ++j) wtCa[i][j] = (__bf16*)carve(65536 * 2);
    wtDecF1[i] = (__bf16*)carve(262144 * 2);
    wtDecF2[i] = (__bf16*)carve(262144 * 2);
  }
  float*  z_f  = (float*) carve((size_t)Me * 256 * 4);
  __bf16* z_b  = (__bf16*)carve((size_t)Me * 256 * 2);
  float*  t_f  = (float*) carve((size_t)Md * 256 * 4);
  __bf16* t_b  = (__bf16*)carve((size_t)Md * 256 * 2);
  __bf16* q_b  = (__bf16*)carve((size_t)Md * 256 * 2);
  __bf16* k_b  = (__bf16*)carve((size_t)Md * 256 * 2);
  __bf16* v_b  = (__bf16*)carve((size_t)Md * 256 * 2);
  __bf16* a_b  = (__bf16*)carve((size_t)Md * 256 * 2);
  float*  proj = (float*) carve((size_t)Md * 256 * 4);
  __bf16* h_b  = (__bf16*)carve((size_t)Md * 1024 * 2);
  __bf16* kv_b = (__bf16*)carve(4 * 8 * 32 * 32 * 2);
  float*  ksum = (float*) carve(4 * 8 * 32 * 4);
  float2* dft  = (float2*)carve((size_t)4 * 8320 * 8);
  float2* tmc  = (float2*)carve((size_t)4 * 128 * 65 * 8);
  float*  img  = (float*) carve((size_t)4 * 16384 * 4);
  float*  cb1  = (float*) carve((size_t)4 * 32 * 16384 * 4);
  float*  cb2  = (float*) carve((size_t)4 * 32 * 16384 * 4);
  float*  mv   = (float*) carve(64 * 4);

  auto cw = [&](const float* W, __bf16* Wt, int K, int N) {
    int n = K * N;
    k_wt<<<(n + 255) / 256, 256, 0, stream>>>(W, Wt, K, N);
  };
  for (int i = 0; i < 4; ++i) {
    for (int j = 0; j < 4; ++j) cw(enc_W + (size_t)(i * 4 + j) * 65536, wtEnc[i][j], 256, 256);
    cw(enc_f1 + (size_t)i * 262144, wtEncF1[i], 256, 1024);
    cw(enc_f2 + (size_t)i * 262144, wtEncF2[i], 1024, 256);
    for (int j = 0; j < 4; ++j) cw(dsa_W + (size_t)(i * 4 + j) * 65536, wtSa[i][j], 256, 256);
    for (int j = 0; j < 4; ++j) cw(dca_W + (size_t)(i * 4 + j) * 65536, wtCa[i][j], 256, 256);
    cw(dec_f1 + (size_t)i * 262144, wtDecF1[i], 256, 1024);
    cw(dec_f2 + (size_t)i * 262144, wtDecF2[i], 1024, 256);
  }

  auto gemm = [&](const __bf16* A, const __bf16* Wt, const float* bias,
                  float* Cf, __bf16* Cb, int M, int N, int K, bool relu) {
    dim3 g(N / 64, M / 128), blk(128);
    if (relu)          k_gemm<true , false, true ><<<g, blk, 0, stream>>>(A, Wt, bias, Cf, Cb, M, N, K);
    else if (Cf && Cb) k_gemm<false, true , true ><<<g, blk, 0, stream>>>(A, Wt, bias, Cf, Cb, M, N, K);
    else if (Cf)       k_gemm<false, true , false><<<g, blk, 0, stream>>>(A, Wt, bias, Cf, Cb, M, N, K);
    else               k_gemm<false, false, true ><<<g, blk, 0, stream>>>(A, Wt, bias, Cf, Cb, M, N, K);
  };

  auto attn = [&](const __bf16* xq, const __bf16* xkv, int Lq, int Lkv,
                  __bf16* const* Wt4, const float* bias4) {
    int Mq = Bz * Lq, Mkv = Bz * Lkv;
    gemm(xq , Wt4[0], bias4 + 0 * 256, nullptr, q_b, Mq , 256, 256, false);
    gemm(xkv, Wt4[1], bias4 + 1 * 256, nullptr, k_b, Mkv, 256, 256, false);
    gemm(xkv, Wt4[2], bias4 + 2 * 256, nullptr, v_b, Mkv, 256, 256, false);
    k_elu1<<<(Mq  * 256) / 256, 256, 0, stream>>>(q_b, Mq  * 256);
    k_elu1<<<(Mkv * 256) / 256, 256, 0, stream>>>(k_b, Mkv * 256);
    k_kv  <<<32, 128, 0, stream>>>(k_b, v_b, kv_b, Lkv);
    k_ksum<<<32, 128, 0, stream>>>(k_b, ksum, Lkv);
    k_attn_out<<<dim3(Lq / 64, 32), 128, 0, stream>>>(q_b, kv_b, ksum, a_b, Lq);
    gemm(a_b, Wt4[3], bias4 + 3 * 256, proj, nullptr, Mq, 256, 256, false);
  };

  auto addln = [&](float* xf, const float* yf, const float* g, const float* b,
                   float* of, __bf16* ob, int M) {
    k_add_ln<<<M / 8, 256, 0, stream>>>(xf, yf, g, b, of, ob, M);
  };

  // --- embedding + encoder ---
  k_embed<<<Me, 256, 0, stream>>>(x, pe_input, emb_W, emb_b, z_f, z_b);
  for (int i = 0; i < 4; ++i) {
    attn(z_b, z_b, Le, Le, wtEnc[i], enc_b + (size_t)i * 4 * 256);
    addln(z_f, proj, enc_lng + (size_t)(i * 2 + 0) * 256, enc_lnb + (size_t)(i * 2 + 0) * 256, z_f, z_b, Me);
    gemm(z_b, wtEncF1[i], enc_f1b + (size_t)i * 1024, nullptr, h_b, Me, 1024, 256, true);
    gemm(h_b, wtEncF2[i], enc_f2b + (size_t)i * 256, proj, nullptr, Me, 256, 1024, false);
    addln(z_f, proj, enc_lng + (size_t)(i * 2 + 1) * 256, enc_lnb + (size_t)(i * 2 + 1) * 256, z_f, z_b, Me);
  }

  // --- decoder ---
  k_bcast<<<(Md * 256) / 256, 256, 0, stream>>>(out_pos, t_f, t_b, Md * 256);
  for (int i = 0; i < 4; ++i) {
    attn(t_b, t_b, Ld, Ld, wtSa[i], dsa_b + (size_t)i * 4 * 256);
    addln(t_f, proj, dec_lng + (size_t)(i * 3 + 0) * 256, dec_lnb + (size_t)(i * 3 + 0) * 256, t_f, t_b, Md);
    attn(t_b, z_b, Ld, Le, wtCa[i], dca_b + (size_t)i * 4 * 256);
    addln(t_f, proj, dec_lng + (size_t)(i * 3 + 1) * 256, dec_lnb + (size_t)(i * 3 + 1) * 256, t_f, t_b, Md);
    gemm(t_b, wtDecF1[i], dec_f1b + (size_t)i * 1024, nullptr, h_b, Md, 1024, 256, true);
    gemm(h_b, wtDecF2[i], dec_f2b + (size_t)i * 256, proj, nullptr, Md, 256, 1024, false);
    addln(t_f, proj, dec_lng + (size_t)(i * 3 + 2) * 256, dec_lnb + (size_t)(i * 3 + 2) * 256, t_f, t_b, Md);
  }

  // --- head + iFFT + conv post-net ---
  k_pred<<<(Md + 255) / 256, 256, 0, stream>>>(t_b, pred_W, pred_b, out, Md);
  k_dft <<<(Md + 255) / 256, 256, 0, stream>>>(out, coords, mag_min, mag_max, dft, Md);
  k_atten<<<(Md + 255) / 256, 256, 0, stream>>>(dft, atten, Md);
  k_ifft_y<<<(4 * 128 * 65 + 255) / 256, 256, 0, stream>>>(dft, tmc);
  k_ifft_x<<<(4 * 128 * 128) / 256, 256, 0, stream>>>(tmc, img);
  k_conv1<<<(1 << 21) / 256, 256, 0, stream>>>(img, c1w, c1b, cb1);
  k_bnstats<<<32, 256, 0, stream>>>(cb1, mv);
  k_bnapply<<<(1 << 21) / 256, 256, 0, stream>>>(cb1, mv, bn1g, bn1b);
  k_conv2<<<(1 << 21) / 256, 256, 0, stream>>>(cb1, c2w, c2b, cb2);
  k_bnstats<<<32, 256, 0, stream>>>(cb2, mv);
  k_bnapply<<<(1 << 21) / 256, 256, 0, stream>>>(cb2, mv, bn2g, bn2b);
  k_conv3<<<(4 * 16384) / 256, 256, 0, stream>>>(cb2, img, c3w, c3b, out + (size_t)Md * 2);
}
